// MultiHeadAttention_82660940579150
// MI455X (gfx1250) — matrix operand order repaired and checked by reading, not compiled, not run on hardware
//
#include <hip/hip_runtime.h>
#include <hip/hip_bf16.h>

typedef __attribute__((ext_vector_type(16))) _Float16 v16h;
typedef __attribute__((ext_vector_type(8)))  _Float16 v8h;
typedef __attribute__((ext_vector_type(4)))  _Float16 v4h;
typedef __attribute__((ext_vector_type(8)))  float    v8f;

#define HSZ 1024
#define NHD 16
#define HD  64
#define BB  2
#define SS  2048
#define MTOT (BB*SS)   // 4096

// ---------------------------------------------------------------------------
// CDNA5 async global->LDS copy (CDNA5 ISA). ASYNCcnt-
// tracked; async loads complete IN ORDER, so with 4 copies per stage,
// "s_wait_asynccnt 4" waits for the older stage while the newer one flies.
// ---------------------------------------------------------------------------
__device__ __forceinline__ void async_b128(unsigned lds_addr, const void* gptr) {
  asm volatile("global_load_async_to_lds_b128 %0, %1, off"
               :: "v"(lds_addr), "v"(gptr) : "memory");
}
__device__ __forceinline__ void wait_async0() {
  asm volatile("s_wait_asynccnt 0" ::: "memory");
}
__device__ __forceinline__ void wait_async4() {
  asm volatile("s_wait_asynccnt 4" ::: "memory");
}
__device__ __forceinline__ unsigned lds_off(const void* p) {
  return (unsigned)(size_t)p;   // generic LDS pointer: addr[31:0] = LDS offset
}

// ---------------------------------------------------------------------------
// Fragment helpers (wave32 WMMA 16x16x32 f16; this code's contiguous k order, same for A and B)
// A 16x32: lane m=lane&15; VGPR0-7 hold K=kA..kA+15 (contiguous),
//          kA = (lane>>4)*16.
// B 32x16: lane n=lane&15 holds column n, K = (lane>>4)*16 .. +15 (contiguous).
// C/D 16x16 f32: VGPR p, lane l -> row p + 8*(l>>4), col l&15.
// ---------------------------------------------------------------------------
__device__ __forceinline__ v16h load_a_frag(const _Float16* row0, int stride, int lane) {
  const int m = lane & 15, kA = (lane >> 4) * 16;
  const _Float16* p = row0 + m * stride + kA;
  v8h lo = *(const v8h*)p;
  v8h hi = *(const v8h*)(p + 8);
  v16h a;
#pragma unroll
  for (int i = 0; i < 8; i++) { a[i] = lo[i]; a[i + 8] = hi[i]; }
  return a;
}

// ---------------------------------------------------------------------------
// One-shot f32 -> f16 conversion (x and the four weight matrices).
// ~30 MB total traffic => ~1.3 us at 23.3 TB/s; keeps cvt out of GEMM loops.
// ---------------------------------------------------------------------------
__global__ __launch_bounds__(256) void cvt_f32_to_f16(const float* __restrict__ src,
                                                      _Float16* __restrict__ dst,
                                                      int n4) {
  int i = blockIdx.x * blockDim.x + threadIdx.x;
  if (i < n4) {
    float4 v = ((const float4*)src)[i];
    v4h o = { (_Float16)v.x, (_Float16)v.y, (_Float16)v.z, (_Float16)v.w };
    ((v4h*)dst)[i] = o;
  }
}

// ---------------------------------------------------------------------------
// Pure-f16 GEMM: out = (A[M,K] * B[N,K]^T + bias) * scale.
// LAYOUT 0 -> f16 out [b,h,s,d] (Q,K)   LAYOUT 1 -> f16 out [b,h,d,s] (V^T)
// LAYOUT 2 -> f32 out row-major [M,N]   (final projection to d_out)
// Block tile 128x128x32, 256 threads = 8 waves, each wave 32x64 (2x4 frags).
// Double-buffered LDS; async copies of panel i+1 overlap WMMA on panel i.
// ---------------------------------------------------------------------------
template<int LAYOUT>
__global__ __launch_bounds__(256) void gemm_f16(const _Float16* __restrict__ A,
                                                const _Float16* __restrict__ Bw,
                                                const float* __restrict__ bias,
                                                void* __restrict__ outp,
                                                float scale)
{
  __shared__ alignas(32) _Float16 As[2][128 * 32];
  __shared__ alignas(32) _Float16 Bs[2][128 * 32];
  const int tid  = threadIdx.x;
  const int lane = tid & 31, wid = tid >> 5;
  const int m0 = blockIdx.y * 128, n0 = blockIdx.x * 128;
  const int wm = (wid & 3) * 32, wn = (wid >> 2) * 64;
  const int lrow = tid >> 1;           // 0..127
  const int lcol = (tid & 1) * 16;     // 0 / 16 halves

  auto stage = [&](int k0, int buf) {  // 4 async b128 per thread per stage
    const _Float16* ga = A  + (size_t)(m0 + lrow) * HSZ + k0 + lcol;
    const _Float16* gb = Bw + (size_t)(n0 + lrow) * HSZ + k0 + lcol;
    const unsigned la = lds_off(&As[buf][lrow * 32 + lcol]);
    const unsigned lb = lds_off(&Bs[buf][lrow * 32 + lcol]);
    async_b128(la,      ga);
    async_b128(la + 16, ga + 8);
    async_b128(lb,      gb);
    async_b128(lb + 16, gb + 8);
  };

  v8f acc[2][4];
#pragma unroll
  for (int i = 0; i < 2; i++)
#pragma unroll
    for (int j = 0; j < 4; j++) acc[i][j] = v8f{};

  stage(0, 0);
  for (int k0 = 0, it = 0; k0 < HSZ; k0 += 32, ++it) {
    const int c = it & 1;
    const bool more = (k0 + 32 < HSZ);
    if (more) stage(k0 + 32, c ^ 1);   // prefetch next panel into other buffer
    if (more) wait_async4(); else wait_async0();   // current panel landed
    __syncthreads();

    // hoist all fragments, then run the 8-WMMA chain back-to-back
    const int nn = lane & 15, kB = (lane >> 4) * 16;
    v16h afr[2], bfr[4];
#pragma unroll
    for (int i = 0; i < 2; i++)
      afr[i] = load_a_frag(&As[c][(wm + i * 16) * 32], 32, lane);
#pragma unroll
    for (int j = 0; j < 4; j++)
      bfr[j] = *(const v16h*)&Bs[c][(wn + j * 16 + nn) * 32 + kB];
#pragma unroll
    for (int i = 0; i < 2; i++)
#pragma unroll
      for (int j = 0; j < 4; j++)
        acc[i][j] = __builtin_amdgcn_wmma_f32_16x16x32_f16(
            false, afr[i], false, bfr[j], (short)0, acc[i][j], false, false);

    __syncthreads();                   // this buffer free for re-staging
  }

  const int half = lane >> 4, cOff = lane & 15;
#pragma unroll
  for (int i = 0; i < 2; i++)
#pragma unroll
    for (int j = 0; j < 4; j++) {
      const int col = n0 + wn + j * 16 + cOff;
      const float bv = bias[col];
      const int h = col >> 6, d = col & 63;
#pragma unroll
      for (int p = 0; p < 8; p++) {
        const int row = m0 + wm + i * 16 + p + half * 8;
        const float v = (acc[i][j][p] + bv) * scale;
        if (LAYOUT == 2) {
          ((float*)outp)[(size_t)row * HSZ + col] = v;
        } else {
          const int b = row >> 11, s = row & (SS - 1);
          size_t idx;
          if (LAYOUT == 0) idx = (((size_t)(b * NHD + h)) * SS + s) * HD + d;
          else             idx = (((size_t)(b * NHD + h)) * HD + d) * SS + s;
          ((_Float16*)outp)[idx] = (_Float16)v;
        }
      }
    }
}

// ---------------------------------------------------------------------------
// Flash attention: grid (S/64, B*NH), 128 threads = 4 waves, one 16-row query
// tile per wave, key steps of 32. K (32x64) and V^T (64x32) panels double-
// buffered in LDS via async copies (next panel overlaps current WMMA work;
// 4x reuse across the 4 waves). Q arrives pre-scaled by 1/sqrt(64) from the
// Q projection, so no score scaling here.
// Q,K in [bh][s][d] f16; V in [bh][d][s] f16; ctx out as [b*S][H] f16.
// ---------------------------------------------------------------------------
__global__ __launch_bounds__(128) void attn_kernel(const _Float16* __restrict__ Q,
                                                   const _Float16* __restrict__ K,
                                                   const _Float16* __restrict__ V,
                                                   _Float16* __restrict__ Ctx)
{
  __shared__ alignas(32) _Float16 Ks[2][32 * 64];   // [key][d]
  __shared__ alignas(32) _Float16 Vs[2][64 * 32];   // [d][key-slice]
  __shared__ alignas(32) _Float16 Pl[4][16 * 32];   // per-wave P scratch
  const int tid = threadIdx.x;
  const int lane = tid & 31, wid = tid >> 5;
  const int bh = blockIdx.y;
  const int b = bh >> 4, h = bh & 15;
  const int q0 = (blockIdx.x * 4 + wid) * 16;
  const _Float16* Qp = Q + ((size_t)bh * SS + q0) * HD;
  const _Float16* Kp = K + (size_t)bh * SS * HD;
  const _Float16* Vt = V + (size_t)bh * HD * SS;

  const int m = lane & 15, half = lane >> 4;
  const int kA = half * 16;

  // staging assignments (128 threads, 4 KB per panel, 32 B per thread each)
  const int krow = tid >> 2, kcol = (tid & 3) * 16;   // K: 32 rows x 64 halves
  const int vrow = tid >> 1, vcol = (tid & 1) * 16;   // V: 64 rows x 32 halves

  auto stage = [&](int kb, int buf) {  // 4 async b128 per thread per stage
    const _Float16* gk = Kp + (size_t)(kb + krow) * HD + kcol;
    const _Float16* gv = Vt + (size_t)vrow * SS + kb + vcol;
    const unsigned lk = lds_off(&Ks[buf][krow * 64 + kcol]);
    const unsigned lv = lds_off(&Vs[buf][vrow * 32 + vcol]);
    async_b128(lk,      gk);
    async_b128(lk + 16, gk + 8);
    async_b128(lv,      gv);
    async_b128(lv + 16, gv + 8);
  };

  v16h aQ[2];
#pragma unroll
  for (int t = 0; t < 2; t++) {
    const _Float16* p = Qp + m * HD + t * 32 + kA;
    v8h lo = *(const v8h*)p, hi = *(const v8h*)(p + 8);
#pragma unroll
    for (int i = 0; i < 8; i++) { aQ[t][i] = lo[i]; aQ[t][i + 8] = hi[i]; }
  }

  v8f o[4];
#pragma unroll
  for (int j = 0; j < 4; j++) o[j] = v8f{};
  float mrow[8], lsum[8];
#pragma unroll
  for (int p = 0; p < 8; p++) { mrow[p] = -3.0e38f; lsum[p] = 0.0f; }

  stage(0, 0);
  for (int kb = 0, it = 0; kb < SS; kb += 32, ++it) {
    const int c = it & 1;
    const bool more = (kb + 32 < SS);
    if (more) stage(kb + 32, c ^ 1);
    if (more) wait_async4(); else wait_async0();
    __syncthreads();

    // ---- scores S = Q_scaled K^T, two 16x16 tiles over 32 keys ----
    v16h bK[2][2];
#pragma unroll
    for (int t = 0; t < 2; t++)
#pragma unroll
      for (int dd = 0; dd < 2; dd++)
        bK[t][dd] = *(const v16h*)&Ks[c][(t * 16 + m) * 64 + dd * 32 + half * 16];
    v8f sc[2];
#pragma unroll
    for (int t = 0; t < 2; t++) {
      v8f acc = v8f{};
#pragma unroll
      for (int dd = 0; dd < 2; dd++)
        acc = __builtin_amdgcn_wmma_f32_16x16x32_f16(false, aQ[dd], false, bK[t][dd],
                                                     (short)0, acc, false, false);
      sc[t] = acc;
    }

    // ---- online softmax per row (row p+8*half; 16 cols live across a half-wave)
#pragma unroll
    for (int p = 0; p < 8; p++) {
      float mx = fmaxf(sc[0][p], sc[1][p]);
#pragma unroll
      for (int off = 8; off; off >>= 1) mx = fmaxf(mx, __shfl_xor(mx, off, 32));
      const float mn = fmaxf(mrow[p], mx);
      const float al = __expf(mrow[p] - mn);
      mrow[p] = mn;
      const float a0 = __expf(sc[0][p] - mn);
      const float a1 = __expf(sc[1][p] - mn);
      float rs = a0 + a1;
#pragma unroll
      for (int off = 8; off; off >>= 1) rs += __shfl_xor(rs, off, 32);
      lsum[p] = lsum[p] * al + rs;
#pragma unroll
      for (int j = 0; j < 4; j++) o[j][p] *= al;
      const int r = p + half * 8;
      Pl[wid][r * 32 + (lane & 15)]      = (_Float16)a0;   // D-layout -> LDS
      Pl[wid][r * 32 + 16 + (lane & 15)] = (_Float16)a1;
    }

    // ---- reload P with A-fragment layout (intra-wave; compiler emits dscnt waits)
    v16h aP;
    {
      const _Float16* p = &Pl[wid][m * 32 + kA];
      v8h lo = *(const v8h*)p, hi = *(const v8h*)(p + 8);
#pragma unroll
      for (int i = 0; i < 8; i++) { aP[i] = lo[i]; aP[i + 8] = hi[i]; }
    }

    // ---- O += P * V  (V^T panel: lane n=d reads 16 contiguous keys) ----
    v16h bV[4];
#pragma unroll
    for (int j = 0; j < 4; j++)
      bV[j] = *(const v16h*)&Vs[c][(j * 16 + m) * 32 + half * 16];
#pragma unroll
    for (int j = 0; j < 4; j++)
      o[j] = __builtin_amdgcn_wmma_f32_16x16x32_f16(false, aP, false, bV[j],
                                                    (short)0, o[j], false, false);

    __syncthreads();                   // this K/V buffer free for re-staging
  }

  // ---- normalize and emit ctx as f16 [B*S, H] for the output projection ----
#pragma unroll
  for (int p = 0; p < 8; p++) {
    const float inv = 1.0f / lsum[p];
    const int s = q0 + p + half * 8;
    const size_t mr = (size_t)b * SS + s;
#pragma unroll
    for (int j = 0; j < 4; j++) {
      const int col = h * HD + j * 16 + (lane & 15);
      Ctx[mr * HSZ + col] = (_Float16)(o[j][p] * inv);
    }
  }
}

// ---------------------------------------------------------------------------
extern "C" void kernel_launch(void* const* d_in, const int* in_sizes, int n_in,
                              void* d_out, int out_size, void* d_ws, size_t ws_size,
                              hipStream_t stream) {
  const float* x  = (const float*)d_in[0];
  const float* Wq = (const float*)d_in[1];
  const float* bq = (const float*)d_in[2];
  const float* Wk = (const float*)d_in[3];
  const float* bk = (const float*)d_in[4];
  const float* Wv = (const float*)d_in[5];
  const float* bv = (const float*)d_in[6];
  const float* Wo = (const float*)d_in[7];
  const float* bo = (const float*)d_in[8];

  _Float16* ws = (_Float16*)d_ws;
  const size_t nx = (size_t)MTOT * HSZ;   // 4M elems
  const size_t nw = (size_t)HSZ * HSZ;    // 1M elems
  _Float16* Xh  = ws;
  _Float16* Wqh = Xh  + nx;
  _Float16* Wkh = Wqh + nw;
  _Float16* Wvh = Wkh + nw;
  _Float16* Woh = Wvh + nw;
  _Float16* Qb  = Woh + nw;
  _Float16* Kb  = Qb  + nx;
  _Float16* Vb  = Kb  + nx;
  _Float16* Cb  = Vb  + nx;               // total 24M halves = 48 MB

  // ---- precision pass: everything to f16 once ----
  cvt_f32_to_f16<<<(int)(nx / 4 + 255) / 256, 256, 0, stream>>>(x,  Xh,  (int)(nx / 4));
  cvt_f32_to_f16<<<(int)(nw / 4 + 255) / 256, 256, 0, stream>>>(Wq, Wqh, (int)(nw / 4));
  cvt_f32_to_f16<<<(int)(nw / 4 + 255) / 256, 256, 0, stream>>>(Wk, Wkh, (int)(nw / 4));
  cvt_f32_to_f16<<<(int)(nw / 4 + 255) / 256, 256, 0, stream>>>(Wv, Wvh, (int)(nw / 4));
  cvt_f32_to_f16<<<(int)(nw / 4 + 255) / 256, 256, 0, stream>>>(Wo, Woh, (int)(nw / 4));

  dim3 ggrid(HSZ / 128, MTOT / 128);      // (8, 32)
  dim3 gblk(256);
  // Q projection pre-scaled by 1/sqrt(HD): softmax(qk/8) == softmax((q/8)k)
  gemm_f16<0><<<ggrid, gblk, 0, stream>>>(Xh, Wqh, bq, Qb, 0.125f);
  gemm_f16<0><<<ggrid, gblk, 0, stream>>>(Xh, Wkh, bk, Kb, 1.0f);
  gemm_f16<1><<<ggrid, gblk, 0, stream>>>(Xh, Wvh, bv, Vb, 1.0f);

  dim3 agrid(SS / 64, BB * NHD);          // (32, 32)
  attn_kernel<<<agrid, dim3(128), 0, stream>>>(Qb, Kb, Vb, Cb);

  gemm_f16<2><<<ggrid, gblk, 0, stream>>>(Cb, Woh, bo, d_out, 1.0f);
}
